// SupervisedInfoNCELoss_79499844649266
// MI455X (gfx1250) — matrix operand order repaired and checked by reading, not compiled, not run on hardware
//
#include <hip/hip_runtime.h>

typedef __attribute__((ext_vector_type(16))) _Float16 v16h;
typedef __attribute__((ext_vector_type(8)))  _Float16 v8h;
typedef __attribute__((ext_vector_type(4)))  _Float16 v4h;
typedef __attribute__((ext_vector_type(8)))  float    v8f;
typedef __attribute__((ext_vector_type(4)))  float    v4f;

#define NROWS   8192
#define DIM     128
#define INV_TAF 20.0f                    // 1 / 0.05; fixed LSE shift (logits <= 20)
#define SCL_L2E 28.853900817779268f      // 20 * log2(e): exp(x*20-20) == exp2(x*SCL - SCL)

// ---------------------------------------------------------------------------
// Kernel 1: row-normalize z, z_p, z_n (fp32) -> packed f16 rows in workspace.
// One wave32 per row: 4 floats/lane, full-wave shfl reduction for ||x||^2.
// Thread (0,0) also zeroes the output accumulator.
// ---------------------------------------------------------------------------
__global__ __launch_bounds__(256) void nce_normalize_f16(
    const float* __restrict__ z, const float* __restrict__ zp,
    const float* __restrict__ znq, _Float16* __restrict__ dst,
    float* __restrict__ out)
{
    if (blockIdx.x == 0 && threadIdx.x == 0) out[0] = 0.0f;

    const int wave = threadIdx.x >> 5;
    const int lane = threadIdx.x & 31;
    const int grow = blockIdx.x * 8 + wave;        // 0 .. 3*NROWS-1

    const float* src;
    if (grow < NROWS)            src = z   + (size_t)grow * DIM;
    else if (grow < 2 * NROWS)   src = zp  + (size_t)(grow - NROWS) * DIM;
    else                         src = znq + (size_t)(grow - 2 * NROWS) * DIM;

    v4f v = *(const v4f*)(src + lane * 4);
    float ss = v.x * v.x + v.y * v.y + v.z * v.z + v.w * v.w;
#pragma unroll
    for (int m = 1; m < 32; m <<= 1) ss += __shfl_xor(ss, m, 32);

    const float inv = 1.0f / fmaxf(sqrtf(ss), 1e-8f);   // torch: x / max(||x||, eps)

    v4h h;
    h.x = (_Float16)(v.x * inv);
    h.y = (_Float16)(v.y * inv);
    h.z = (_Float16)(v.z * inv);
    h.w = (_Float16)(v.w * inv);
    *(v4h*)(dst + (size_t)grow * DIM + lane * 4) = h;
}

// ---------------------------------------------------------------------------
// Kernel 2: streaming InfoNCE. One block per 16-row tile; 8 waves split the
// 1024 column tiles. z_p/z_n rows are CONTIGUOUS in the workspace, so the
// column space is one flat array `cols` and the hot loop advances a single
// pointer by a constant stride. Per tile: 8x global_load_b128, 4x
// v_wmma_f32_16x16x32_f16 (K=128), 4x v_pk_fma_f32 + 8x v_exp_f32 for the
// fixed-shift online sum-of-exp. Diagonal tile recomputed once by wave 0.
// ---------------------------------------------------------------------------
__global__ __launch_bounds__(256) void nce_main(
    const _Float16* __restrict__ zn, const _Float16* __restrict__ cols,
    float* __restrict__ out)
{
    __shared__ float sArr[8][16];     // per-wave, per-row sum of exp
    __shared__ float diagArr[16];     // diagonal logits for this row tile

    const int wave    = threadIdx.x >> 5;
    const int lane    = threadIdx.x & 31;
    const int rowTile = blockIdx.x;                 // 0..511

    // ---- A fragments (16x32 f16 each), held in VGPRs for the whole loop ----
    // A 16x32 layout (matches B): M = lane & 15; lane<16 -> K 0..15,
    // lane>=16 -> K 16..31. Two contiguous 16B chunks per fragment.
    const int M    = rowTile * 16 + (lane & 15);
    const int kOff = (lane < 16) ? 0 : 16;
    const _Float16* arow = zn + (size_t)M * DIM;

    v16h a[4];
#pragma unroll
    for (int kt = 0; kt < 4; ++kt) {
        v8h lo = *(const v8h*)(arow + kt * 32 + kOff);
        v8h hi = *(const v8h*)(arow + kt * 32 + kOff + 8);
#pragma unroll
        for (int i = 0; i < 8; ++i) { a[kt][i] = lo[i]; a[kt][8 + i] = hi[i]; }
    }

    // ---- B fragment addressing (16-bit B 32x16, contiguous K order as for A): N = lane & 15,
    // lanes 0-15 hold K 0..15, lanes 16-31 hold K 16..31 (32B contiguous). ----
    const int nc    = lane & 15;
    const int bkOff = (lane < 16) ? 0 : 16;

    // Column tile j = t*8 + wave; element offset = (j*16 + nc)*DIM + bkOff.
    // Steady-state: advance by 8 tiles = 8*16*DIM halfs per iteration.
    const size_t tileStride = (size_t)8 * 16 * DIM;
    const _Float16* brow = cols + (size_t)(wave * 16 + nc) * DIM + bkOff;

    float s[8];
#pragma unroll
    for (int v = 0; v < 8; ++v) s[v] = 0.0f;

    for (int t = 0; t < 128; ++t) {
        __builtin_prefetch(brow + tileStride, 0, 1);   // global_prefetch_b8

        v8f c = {};
#pragma unroll
        for (int kt = 0; kt < 4; ++kt) {
            v16h b = *(const v16h*)(brow + kt * 32);   // 32B aligned
            c = __builtin_amdgcn_wmma_f32_16x16x32_f16(
                    false, a[kt], false, b, (short)0, c, false, false);
        }
        // C layout: VGPR v holds row M = v + 8*(lane>=16), col N = lane & 15.
        // exp(logit - 20) == exp2(c * (20*log2e) - 20*log2e): one FMA + exp.
#pragma unroll
        for (int v = 0; v < 8; ++v)
            s[v] += __builtin_amdgcn_exp2f(c[v] * SCL_L2E - SCL_L2E);

        brow += tileStride;
    }

    // Reduce s across the 16 lanes that share the same row set.
#pragma unroll
    for (int v = 0; v < 8; ++v) {
#pragma unroll
        for (int m = 1; m < 16; m <<= 1) s[v] += __shfl_xor(s[v], m, 32);
    }
    if (lane == 0) {
#pragma unroll
        for (int v = 0; v < 8; ++v) sArr[wave][v] = s[v];       // rows 0..7
    }
    if (lane == 16) {
#pragma unroll
        for (int v = 0; v < 8; ++v) sArr[wave][8 + v] = s[v];   // rows 8..15
    }

    // Wave 0 recomputes the single diagonal tile (j == rowTile, positive
    // block) and extracts the M == N elements of the C fragment.
    if (wave == 0) {
        const _Float16* drow =
            cols + (size_t)(rowTile * 16 + nc) * DIM + bkOff;
        v8f c = {};
#pragma unroll
        for (int kt = 0; kt < 4; ++kt) {
            v16h b = *(const v16h*)(drow + kt * 32);
            c = __builtin_amdgcn_wmma_f32_16x16x32_f16(
                    false, a[kt], false, b, (short)0, c, false, false);
        }
        float dv = 0.0f;
#pragma unroll
        for (int v = 0; v < 8; ++v)
            if (lane == v || lane == 24 + v) dv = c[v] * INV_TAF;
        if (lane < 8)   diagArr[lane]      = dv;   // rows 0..7
        if (lane >= 24) diagArr[lane - 16] = dv;   // rows 8..15
    }
    __syncthreads();

    if (wave == 0) {
        float loss = 0.0f;
        if (lane < 16) {
            float st = 0.0f;
#pragma unroll
            for (int w = 0; w < 8; ++w) st += sArr[w][lane];
            // per-row: lse - diag = (20 + log(sum exp(logit-20))) - diag
            loss = (INV_TAF + __logf(st)) - diagArr[lane];
        }
#pragma unroll
        for (int m = 1; m < 16; m <<= 1) loss += __shfl_xor(loss, m, 32);
        if (lane == 0) atomicAdd(out, loss * (1.0f / NROWS));
    }
}

// ---------------------------------------------------------------------------
extern "C" void kernel_launch(void* const* d_in, const int* in_sizes, int n_in,
                              void* d_out, int out_size, void* d_ws, size_t ws_size,
                              hipStream_t stream) {
    const float* z  = (const float*)d_in[0];
    const float* zp = (const float*)d_in[1];
    const float* zq = (const float*)d_in[2];
    float* out = (float*)d_out;

    // workspace: 3 * 8192 * 128 f16 = 6 MB of normalized rows.
    // Layout: [zn | zpn | znn] -- zpn & znn contiguous => flat column space.
    _Float16* norm = (_Float16*)d_ws;

    // 3*8192 rows, 8 rows per 256-thread block
    nce_normalize_f16<<<(3 * NROWS) / 8, 256, 0, stream>>>(z, zp, zq, norm, out);

    const _Float16* zn   = norm;
    const _Float16* cols = norm + (size_t)NROWS * DIM;  // [z_p rows ; z_n rows]
    nce_main<<<NROWS / 16, 256, 0, stream>>>(zn, cols, out);
}
